// AttentionBlockTkgTorchRef_249108103232
// MI455X (gfx1250) — matrix operand order repaired and checked by reading, not compiled, not run on hardware
//
#include <hip/hip_runtime.h>
#include <math.h>

// ---------------------------------------------------------------------------
// Problem constants (from setup_inputs): B=32, S=1, H=4096, N=32, D=128, SMAX=8192
// ---------------------------------------------------------------------------
#define B_     32
#define H_     4096
#define NH     32
#define D_     128
#define SMAX_  8192
#define E_     ((NH + 2) * D_)           // 4352
#define SPLITS 16                        // context splits for flash-decode
#define ROWS_PER_SPLIT (SMAX_ / SPLITS)  // 512
#define ATTN_ITERS (ROWS_PER_SPLIT / 32) // 16
#define KSPLIT 8                         // K-split for the two weight GEMMs

typedef __bf16 bf16_t;
typedef __attribute__((ext_vector_type(16))) __bf16 v16bf;
typedef __attribute__((ext_vector_type(8)))  float  v8f;

__device__ __forceinline__ v8f wmma_bf16(v16bf a, v16bf b, v8f c) {
  // (neg_a, A, neg_b, B, c_mod, C, reuse_a, reuse_b) -> v_wmma_f32_16x16x32_bf16
  return __builtin_amdgcn_wmma_f32_16x16x32_bf16(false, a, false, b, (short)0, c,
                                                 false, false);
}

// LDS byte offset of a __shared__ object (flat LDS-aperture addr, low 32 bits)
__device__ __forceinline__ unsigned lds_off32(const void* p) {
  return (unsigned)(unsigned long long)p;
}

// ---------------------------------------------------------------------------
// Workspace layout (bytes, all 256B aligned)
// ---------------------------------------------------------------------------
enum : size_t {
  XN_OFF   = 0,                                              // Xn bf16  [32 x 4096]
  QKV_OFF  = XN_OFF   + (size_t)B_ * H_ * 2,                 // QKV f32  [32 x 4352]
  QBF_OFF  = QKV_OFF  + (size_t)B_ * E_ * 4,                 // Q bf16   (scaled, post-rope)
  QF_OFF   = QBF_OFF  + (size_t)B_ * NH * D_ * 2,            // Q f32    (scaled, post-rope)
  KACT_OFF = QF_OFF   + (size_t)B_ * NH * D_ * 4,            // K active f32 [32 x 128]
  VACT_OFF = KACT_OFF + (size_t)B_ * D_ * 4,                 // V active f32 [32 x 128]
  MP_OFF   = VACT_OFF + (size_t)B_ * D_ * 4,                 // m partial [B x SPLITS x N]
  LP_OFF   = MP_OFF   + (size_t)B_ * SPLITS * NH * 4,        // l partial
  OP_OFF   = LP_OFF   + (size_t)B_ * SPLITS * NH * 4,        // O partial [B x SPLITS x N x D]
  ATTN_OFF = OP_OFF   + (size_t)B_ * SPLITS * NH * D_ * 4,   // attn bf16 [32 x 4096]
  WS_NEED  = ATTN_OFF + (size_t)B_ * H_ * 2
};

// ---------------------------------------------------------------------------
// Kernel 1: RMSNorm over H=4096, emit bf16 activations
// ---------------------------------------------------------------------------
__global__ void k_rmsnorm_x(const float* __restrict__ X, const float* __restrict__ gamma,
                            bf16_t* __restrict__ Xn) {
  int b = blockIdx.x, t = threadIdx.x;            // 256 threads = 8 waves
  const float* x = X + (size_t)b * H_;
  float ss = 0.f;
  for (int i = t; i < H_; i += 256) { float v = x[i]; ss += v * v; }
  for (int off = 16; off > 0; off >>= 1) ss += __shfl_xor(ss, off, 32);
  __shared__ float red[8];
  int wave = t >> 5, lane = t & 31;
  if (lane == 0) red[wave] = ss;
  __syncthreads();
  float tot = 0.f;
  #pragma unroll
  for (int w = 0; w < 8; ++w) tot += red[w];
  float r = rsqrtf(tot / (float)H_ + 1e-6f);
  for (int i = t; i < H_; i += 256)
    Xn[(size_t)b * H_ + i] = (bf16_t)(x[i] * r * gamma[i]);
}

// ---------------------------------------------------------------------------
// Kernel 2: split-K GEMM  C[32 x Eout] (+=) A_bf16[32 x 4096] * W_f32[4096 x Eout]
// One wave per 32x16 output tile slice; KSPLIT blocks accumulate via f32 atomics.
// ---------------------------------------------------------------------------
__global__ void __launch_bounds__(32) k_gemm_rowmajor(
    const bf16_t* __restrict__ A,           // bf16, lda = 4096
    const float* __restrict__ W,            // 4096 x Eout
    float* __restrict__ C,                  // 32 x Eout (pre-zeroed, atomicAdd)
    int Eout) {
  int lane = threadIdx.x;
  int col0 = blockIdx.x * 16;
  int kBeg = blockIdx.y * (H_ / KSPLIT);
  int kEnd = kBeg + (H_ / KSPLIT);
  int lhi = lane >> 4, llo = lane & 15;

  __shared__ __align__(16) bf16_t As[32 * 32];
  v8f acc0 = {}, acc1 = {};

  for (int k0 = kBeg; k0 < kEnd; k0 += 32) {
    // stage A (rows 0..31, 32 k's) to LDS, one 64B row per lane
    const uint4* src = (const uint4*)(A + (size_t)lane * H_ + k0);
    uint4* dst = (uint4*)As;
    #pragma unroll
    for (int j = 0; j < 4; ++j) dst[lane * 4 + j] = src[j];
    __syncthreads();

    if (k0 + 32 < kEnd)
      __builtin_prefetch(&W[(size_t)(k0 + 32) * Eout + col0 + llo], 0, 1);

    // B fragment: k = i + 16*lhi, n = llo  (hardware f32->bf16 cvt)
    v16bf bfr;
    #pragma unroll
    for (int i = 0; i < 16; ++i)
      bfr[i] = (bf16_t)W[(size_t)(k0 + i + (lhi << 4)) * Eout + col0 + llo];

    // A fragments (two 16-row M tiles): k = i + lhi*16
    v16bf a0, a1;
    #pragma unroll
    for (int i = 0; i < 16; ++i) {
      int km = i + (lhi << 4);
      a0[i] = As[(llo) * 32 + km];
      a1[i] = As[(16 + llo) * 32 + km];
    }
    acc0 = wmma_bf16(a0, bfr, acc0);
    acc1 = wmma_bf16(a1, bfr, acc1);
    __syncthreads();
  }
  #pragma unroll
  for (int r = 0; r < 8; ++r) {
    int row = r + (lhi << 3);
    atomicAdd(&C[(size_t)row * Eout + col0 + llo], acc0[r]);
    atomicAdd(&C[(size_t)(row + 16) * Eout + col0 + llo], acc1[r]);
  }
}

// ---------------------------------------------------------------------------
// block reduce over 128 threads (4 waves)
// ---------------------------------------------------------------------------
__device__ __forceinline__ float blockReduceSum128(float v, float* red) {
  for (int off = 16; off > 0; off >>= 1) v += __shfl_xor(v, off, 32);
  int wave = threadIdx.x >> 5, lane = threadIdx.x & 31;
  if (lane == 0) red[wave] = v;
  __syncthreads();
  float t = red[0] + red[1] + red[2] + red[3];
  __syncthreads();
  return t;
}

// ---------------------------------------------------------------------------
// Kernel 3: Q/K RMSNorm + RoPE, fold 1/sqrt(D) into Q, scatter K/V cache rows
// ---------------------------------------------------------------------------
__global__ void k_prep(const float* __restrict__ QKV,
                       const float* __restrict__ gq, const float* __restrict__ gk,
                       const float* __restrict__ cosp, const float* __restrict__ sinp,
                       const long long* __restrict__ kvidx,
                       float* __restrict__ Kout, float* __restrict__ Vout,
                       float* __restrict__ Kact, float* __restrict__ Vact,
                       bf16_t* __restrict__ Qbf, float* __restrict__ Qf) {
  int b = blockIdx.x, t = threadIdx.x;            // 128 threads, t = d index
  __shared__ float red[4];
  __shared__ float xs[128];
  const float scale = 0.08838834764831845f;       // 1/sqrt(128)
  int pos = (int)kvidx[b];

  { // ---- K: rmsnorm + rope, write cache row + active copy
    float x  = QKV[(size_t)b * E_ + NH * D_ + t];
    float ss = blockReduceSum128(x * x, red);
    float r  = rsqrtf(ss / (float)D_ + 1e-6f);
    xs[t] = x * r * gk[t];
    __syncthreads();
    float out;
    if (t < 64) out = xs[t] * cosp[t * B_ + b] - xs[t + 64] * sinp[t * B_ + b];
    else { int d2 = t - 64; out = xs[t] * cosp[d2 * B_ + b] + xs[t - 64] * sinp[d2 * B_ + b]; }
    Kout[((size_t)b * SMAX_ + pos) * D_ + t] = out;
    Kact[(size_t)b * D_ + t] = out;
    __syncthreads();
  }
  { // ---- V: passthrough
    float v = QKV[(size_t)b * E_ + (NH + 1) * D_ + t];
    Vout[((size_t)b * SMAX_ + pos) * D_ + t] = v;
    Vact[(size_t)b * D_ + t] = v;
  }
  // ---- Q heads
  for (int n = 0; n < NH; ++n) {
    float x  = QKV[(size_t)b * E_ + n * D_ + t];
    float ss = blockReduceSum128(x * x, red);
    float r  = rsqrtf(ss / (float)D_ + 1e-6f);
    xs[t] = x * r * gq[t];
    __syncthreads();
    float out;
    if (t < 64) out = xs[t] * cosp[t * B_ + b] - xs[t + 64] * sinp[t * B_ + b];
    else { int d2 = t - 64; out = xs[t] * cosp[d2 * B_ + b] + xs[t - 64] * sinp[d2 * B_ + b]; }
    out *= scale;
    Qf [((size_t)b * NH + n) * D_ + t] = out;
    Qbf[((size_t)b * NH + n) * D_ + t] = (bf16_t)out;
    __syncthreads();
  }
}

// ---------------------------------------------------------------------------
// Kernel 4: flash-decode over the prior cache. One wave per (split, batch).
// Scores:  [32 heads x 128] x [128 x 32 ctx]  (16 WMMAs / 32-ctx tile)
// P*V:     [32 x 32] x [32 x 128]             (16 WMMAs / tile)
// V tiles are double-buffered in LDS via GLOBAL_LOAD_ASYNC_TO_LDS_B128,
// gated with s_wait_asynccnt (32 newest outstanding = next tile in flight).
// ---------------------------------------------------------------------------
__global__ void __launch_bounds__(32) k_attn(
    const bf16_t* __restrict__ Qbf,
    const float* __restrict__ Kc, const float* __restrict__ Vc,
    const unsigned char* __restrict__ mask,
    float* __restrict__ Opart, float* __restrict__ Mpart, float* __restrict__ Lpart) {
  int split = blockIdx.x, b = blockIdx.y;
  int lane = threadIdx.x;
  int lhi = lane >> 4, llo = lane & 15;
  __shared__ __align__(16) float  Vlds[2][32 * D_];   // 2 x 16KB double buffer
  __shared__ __align__(16) bf16_t Pl[32 * 32];        // P repack (D-frag -> A-frag)

  // Q A-fragments [mtile][kchunk], resident for the whole scan
  v16bf qf[2][4];
  #pragma unroll
  for (int m = 0; m < 2; ++m)
    #pragma unroll
    for (int kc = 0; kc < 4; ++kc)
      #pragma unroll
      for (int i = 0; i < 16; ++i) {
        int km = kc * 32 + i + (lhi << 4);
        qf[m][kc][i] = Qbf[((size_t)b * NH + m * 16 + llo) * D_ + km];
      }

  v8f O[2][8];
  #pragma unroll
  for (int m = 0; m < 2; ++m)
    #pragma unroll
    for (int n = 0; n < 8; ++n) O[m][n] = {};
  float mrow[2][8], lrow[2][8];
  #pragma unroll
  for (int m = 0; m < 2; ++m)
    #pragma unroll
    for (int r = 0; r < 8; ++r) { mrow[m][r] = -3.0e38f; lrow[m][r] = 0.f; }

  int base = split * ROWS_PER_SPLIT;

  // async V-tile stage: lane l copies 16B chunk j of row l (coalesced 512B/issue)
  auto issueV = [&](int rows0, int buf) {
    const char* g0 = (const char*)(Vc + ((size_t)b * SMAX_ + rows0) * D_) + lane * 512;
    unsigned l0 = lds_off32(&Vlds[buf][0]) + lane * 512;
    #pragma unroll
    for (int j = 0; j < 32; ++j) {
      const void* gp = g0 + j * 16;
      unsigned lp = l0 + j * 16;
      asm volatile("global_load_async_to_lds_b128 %0, %1, off"
                   :: "v"(lp), "v"(gp) : "memory");
    }
  };

  issueV(base, 0);                                   // prefetch tile 0

  for (int it = 0; it < ATTN_ITERS; ++it) {
    int rows0 = base + it * 32;
    int buf = it & 1;

    // prefetch next V tile, then wait until only those 32 remain outstanding
    if (it + 1 < ATTN_ITERS) {
      issueV(rows0 + 32, buf ^ 1);
      asm volatile("s_wait_asynccnt 0x20" ::: "memory");
    } else {
      asm volatile("s_wait_asynccnt 0x0" ::: "memory");
    }

    // ---- scores: Q[32x128] x K^T[128x32]
    v8f S[2][2]; S[0][0] = {}; S[0][1] = {}; S[1][0] = {}; S[1][1] = {};
    #pragma unroll
    for (int kc = 0; kc < 4; ++kc) {
      #pragma unroll
      for (int nt = 0; nt < 2; ++nt) {
        v16bf kf;   // B-frag: col = ctx row, k = kc*32 + i + 16*lhi (64B vectorized)
        const float4* kp = (const float4*)(Kc + ((size_t)b * SMAX_ + rows0 + nt * 16 + llo) * D_
                                              + kc * 32 + (lhi << 4));
        #pragma unroll
        for (int j = 0; j < 4; ++j) {
          float4 f4 = kp[j];
          kf[j * 4 + 0] = (bf16_t)f4.x; kf[j * 4 + 1] = (bf16_t)f4.y;
          kf[j * 4 + 2] = (bf16_t)f4.z; kf[j * 4 + 3] = (bf16_t)f4.w;
        }
        S[0][nt] = wmma_bf16(qf[0][kc], kf, S[0][nt]);
        S[1][nt] = wmma_bf16(qf[1][kc], kf, S[1][nt]);
      }
    }

    // ---- online softmax: head h's 32 ctx scores live across one 16-lane half
    float alpha[2][8];
    #pragma unroll
    for (int m = 0; m < 2; ++m) {
      #pragma unroll
      for (int r = 0; r < 8; ++r) {
        int head = m * 16 + r + (lhi << 3);
        float sv0 = S[m][0][r], sv1 = S[m][1][r];
        int ctx0 = rows0 + llo, ctx1 = rows0 + 16 + llo;
        if (!mask[((size_t)ctx0 * B_ + b) * NH + head]) sv0 = -30000.f;
        if (!mask[((size_t)ctx1 * B_ + b) * NH + head]) sv1 = -30000.f;
        float mx = fmaxf(sv0, sv1);
        for (int off = 8; off > 0; off >>= 1) mx = fmaxf(mx, __shfl_xor(mx, off, 16));
        float mnew = fmaxf(mrow[m][r], mx);
        float a  = __expf(mrow[m][r] - mnew);
        float p0 = __expf(sv0 - mnew), p1 = __expf(sv1 - mnew);
        float ps = p0 + p1;
        for (int off = 8; off > 0; off >>= 1) ps += __shfl_xor(ps, off, 16);
        lrow[m][r] = lrow[m][r] * a + ps;
        mrow[m][r] = mnew;
        alpha[m][r] = a;
        Pl[head * 32 + llo]      = (bf16_t)p0;
        Pl[head * 32 + 16 + llo] = (bf16_t)p1;
      }
    }
    __syncthreads();

    // ---- rescale O accumulators
    #pragma unroll
    for (int m = 0; m < 2; ++m)
      #pragma unroll
      for (int n = 0; n < 8; ++n)
        #pragma unroll
        for (int r = 0; r < 8; ++r) O[m][n][r] *= alpha[m][r];

    // ---- P A-fragments from LDS
    v16bf pf[2];
    #pragma unroll
    for (int m = 0; m < 2; ++m)
      #pragma unroll
      for (int i = 0; i < 16; ++i) {
        int km = i + (lhi << 4);
        pf[m][i] = Pl[(m * 16 + llo) * 32 + km];
      }

    // ---- O += P * V   (V B-frags gathered from async-staged LDS tile)
    #pragma unroll
    for (int n = 0; n < 8; ++n) {
      v16bf vf;
      #pragma unroll
      for (int i = 0; i < 16; ++i) {
        int row = i + (lhi << 4);                 // ctx row within tile
        vf[i] = (bf16_t)Vlds[buf][row * D_ + n * 16 + llo];
      }
      O[0][n] = wmma_bf16(pf[0], vf, O[0][n]);
      O[1][n] = wmma_bf16(pf[1], vf, O[1][n]);
    }
    __syncthreads();
  }

  // ---- write partials
  #pragma unroll
  for (int m = 0; m < 2; ++m)
    #pragma unroll
    for (int n = 0; n < 8; ++n)
      #pragma unroll
      for (int r = 0; r < 8; ++r) {
        int head = m * 16 + r + (lhi << 3);
        Opart[(((size_t)b * SPLITS + split) * NH + head) * D_ + n * 16 + llo] = O[m][n][r];
      }
  if (llo == 0) {
    #pragma unroll
    for (int m = 0; m < 2; ++m)
      #pragma unroll
      for (int r = 0; r < 8; ++r) {
        int head = m * 16 + r + (lhi << 3);
        Mpart[((size_t)b * SPLITS + split) * NH + head] = mrow[m][r];
        Lpart[((size_t)b * SPLITS + split) * NH + head] = lrow[m][r];
      }
  }
}

// ---------------------------------------------------------------------------
// Kernel 5: combine splits + active token, emit attn (bf16) for out-projection
// ---------------------------------------------------------------------------
__global__ void k_reduce(const float* __restrict__ Qf, const float* __restrict__ Kact,
                         const float* __restrict__ Vact,
                         const float* __restrict__ Opart, const float* __restrict__ Mpart,
                         const float* __restrict__ Lpart,
                         bf16_t* __restrict__ attn) {
  int h = blockIdx.x, b = blockIdx.y, t = threadIdx.x;   // 128 threads
  __shared__ float red[4];
  float qv = Qf[((size_t)b * NH + h) * D_ + t];
  float kv = Kact[(size_t)b * D_ + t];
  float s_act = blockReduceSum128(qv * kv, red);          // Q already 1/sqrt(D)-scaled

  float mg = s_act;
  for (int i = 0; i < SPLITS; ++i)
    mg = fmaxf(mg, Mpart[((size_t)b * SPLITS + i) * NH + h]);

  float acc = 0.f, lsum = 0.f;
  for (int i = 0; i < SPLITS; ++i) {
    float w = __expf(Mpart[((size_t)b * SPLITS + i) * NH + h] - mg);
    lsum += Lpart[((size_t)b * SPLITS + i) * NH + h] * w;
    acc  += w * Opart[(((size_t)b * SPLITS + i) * NH + h) * D_ + t];
  }
  float pa = __expf(s_act - mg);
  lsum += pa;
  acc  += pa * Vact[(size_t)b * D_ + t];
  attn[(size_t)b * H_ + h * D_ + t] = (bf16_t)(acc / lsum);
}

// ---------------------------------------------------------------------------
// Host launcher
// ---------------------------------------------------------------------------
extern "C" void kernel_launch(void* const* d_in, const int* in_sizes, int n_in,
                              void* d_out, int out_size, void* d_ws, size_t ws_size,
                              hipStream_t stream) {
  (void)in_sizes; (void)n_in; (void)out_size; (void)ws_size;

  const float* X     = (const float*)d_in[0];
  const float* gX    = (const float*)d_in[1];
  const float* Wqkv  = (const float*)d_in[2];
  const float* gq    = (const float*)d_in[3];
  const float* gk    = (const float*)d_in[4];
  const float* cosp  = (const float*)d_in[5];
  const float* sinp  = (const float*)d_in[6];
  const float* Kc    = (const float*)d_in[7];
  const float* Vc    = (const float*)d_in[8];
  const float* Wout  = (const float*)d_in[9];
  const unsigned char* mask = (const unsigned char*)d_in[10];
  const long long* kvidx    = (const long long*)d_in[11];

  float* Xout = (float*)d_out;
  float* Kout = Xout + (size_t)B_ * H_;
  float* Vout = Kout + (size_t)B_ * SMAX_ * D_;

  char* ws = (char*)d_ws;
  bf16_t* XN   = (bf16_t*)(ws + XN_OFF);
  float*  QKVw = (float*)(ws + QKV_OFF);
  bf16_t* QBF  = (bf16_t*)(ws + QBF_OFF);
  float*  QF   = (float*)(ws + QF_OFF);
  float*  KACT = (float*)(ws + KACT_OFF);
  float*  VACT = (float*)(ws + VACT_OFF);
  float*  MP   = (float*)(ws + MP_OFF);
  float*  LP   = (float*)(ws + LP_OFF);
  float*  OP   = (float*)(ws + OP_OFF);
  bf16_t* ATTN = (bf16_t*)(ws + ATTN_OFF);

  // 1) outputs include full caches: d2d copy, updated rows overwritten by k_prep
  size_t cacheBytes = (size_t)B_ * SMAX_ * D_ * sizeof(float);
  hipMemcpyAsync(Kout, Kc, cacheBytes, hipMemcpyDeviceToDevice, stream);
  hipMemcpyAsync(Vout, Vc, cacheBytes, hipMemcpyDeviceToDevice, stream);

  // 2) zero atomic-accumulation targets
  hipMemsetAsync(QKVw, 0, (size_t)B_ * E_ * sizeof(float), stream);
  hipMemsetAsync(Xout, 0, (size_t)B_ * H_ * sizeof(float), stream);

  // 3) pipeline
  k_rmsnorm_x<<<B_, 256, 0, stream>>>(X, gX, XN);
  k_gemm_rowmajor<<<dim3(E_ / 16, KSPLIT), 32, 0, stream>>>(XN, Wqkv, QKVw, E_);
  k_prep<<<B_, 128, 0, stream>>>(QKVw, gq, gk, cosp, sinp, kvidx,
                                 Kout, Vout, KACT, VACT, QBF, QF);
  k_attn<<<dim3(SPLITS, B_), 32, 0, stream>>>(QBF, Kc, Vc, mask, OP, MP, LP);
  k_reduce<<<dim3(NH, B_), 128, 0, stream>>>(QF, KACT, VACT, OP, MP, LP, ATTN);
  k_gemm_rowmajor<<<dim3(H_ / 16, KSPLIT), 32, 0, stream>>>(ATTN, Wout, Xout, H_);
}
